// LSTMv1_17214228922509
// MI455X (gfx1250) — matrix operand order repaired and checked by reading, not compiled, not run on hardware
//
#include <hip/hip_runtime.h>
#include <hip/hip_bf16.h>

// ---------------------------------------------------------------------------
// MI455X (gfx1250) persistent LSTM-stack kernel, v2.
//  * Weights cast fp32->bf16 once per launch (134MB -> L2-resident, 192MB L2)
//  * One persistent kernel runs all 512 timesteps x 4 blocks; phases separated
//    by a device-wide monotonic atomic barrier (64 WGs, co-resident)
//  * 512 threads/WG -> 16 wave32/WG -> 4 waves/SIMD for L2 latency hiding
//  * All GEMMs via v_wmma_f32_16x16x32_bf16, double-buffered k-loop fragments
//  * Proj GEMM (K=4096) split 4-way over K; partials combined in a fixed
//    order next phase (deterministic, no float atomics)
// ---------------------------------------------------------------------------

typedef unsigned short u16;
typedef __attribute__((ext_vector_type(16))) __bf16    v16bf;
typedef __attribute__((ext_vector_type(8)))  float     v8f;
typedef __attribute__((ext_vector_type(4)))  unsigned  uq4;
typedef __attribute__((ext_vector_type(4)))  u16       u16x4;
typedef __attribute__((ext_vector_type(4)))  float     f32x4;

union V16U { uq4 q[2]; v16bf v; };

#define NB_    4
#define B_     64
#define S_     512
#define H_     1024
#define H4_    4096
#define NWG_   64
#define NTHR_  512          // 16 wave32 per WG

// round-to-nearest-even fp32 -> bf16
__device__ __forceinline__ u16 f2bf(float f) {
  unsigned u = __float_as_uint(f);
  return (u16)((u + 0x7FFFu + ((u >> 16) & 1u)) >> 16);
}
__device__ __forceinline__ float sigf(float x)  { return 1.0f / (1.0f + expf(-x)); }
__device__ __forceinline__ float geluf(float x) { return 0.5f * x * (1.0f + erff(x * 0.70710678118654752f)); }

// A fragment: X[M,K] row-major, 16x32 bf16 tile.
// Lane l: row m = l&15; half h = l>>4; elems 0..7 = K[h*8..+7], 8..15 = K[16+h*8..+7].
__device__ __forceinline__ v16bf ldA(const u16* X, int ldx, int m0, int k0, int lane) {
  const int m = lane & 15, hf = lane >> 4;
  const u16* p = X + (size_t)(m0 + m) * ldx + k0 + hf * 8;
  V16U u; u.q[0] = *(const uq4*)p; u.q[1] = *(const uq4*)(p + 16);
  return u.v;
}
// B fragment from W[N,K] row-major (computing X@W^T), 32x16 bf16 tile.
// Lane l: col n = l&15; half h = l>>4; elems 0..7 = K[h*8..+7], 8..15 = K[16+h*8..+7].
__device__ __forceinline__ v16bf ldB(const u16* W, int ldw, int n0, int k0, int lane) {
  const int n = lane & 15, hf = lane >> 4;
  const u16* p = W + (size_t)(n0 + n) * ldw + k0 + hf * 8;
  V16U u; u.q[0] = *(const uq4*)p; u.q[1] = *(const uq4*)(p + 16);
  return u.v;
}
__device__ __forceinline__ v8f wmma_bf16(v16bf a, v16bf b, v8f c) {
  return __builtin_amdgcn_wmma_f32_16x16x32_bf16(false, a, false, b, (short)0, c, false, false);
}

// Double-buffered streaming K-loop: loads for step k+1 issue before the WMMA
// of step k, so each wmma's s_wait covers loads issued a full iteration ago.
__device__ __forceinline__ void gemm_k(v8f& acc, const u16* __restrict__ X, int ldx,
                                       const u16* __restrict__ W, int ldw,
                                       int m0, int n0, int kbeg, int klen, int lane) {
  v16bf a = ldA(X, ldx, m0, kbeg, lane);
  v16bf b = ldB(W, ldw, n0, kbeg, lane);
  const int kend = kbeg + klen;
  for (int k0 = kbeg; k0 < kend; k0 += 32) {
    const int kn = (k0 + 32 < kend) ? (k0 + 32) : k0;   // last iter: harmless reload
    v16bf a2 = ldA(X, ldx, m0, kn, lane);
    v16bf b2 = ldB(W, ldw, n0, kn, lane);
    acc = wmma_bf16(a, b, acc);
    a = a2; b = b2;
  }
}

struct Params {
  const int   *stage, *egoc, *oppc, *egoa, *oppa;
  const float *gst, *stab, *ctab, *atab;
  const float *ln1g, *ln1b, *bih, *bhh, *ln2g, *ln2b, *bfc, *bproj;
  const u16   *Wih, *Whh, *Wfc, *Wpr;   // bf16 weights (L2-resident)
  float       *x, *h, *c, *gates, *part, *out;
  u16         *xnbf, *hbf, *fcbf;
  unsigned    *cnt;
};

// Device-wide barrier: monotonic counter, never reset (no reuse hazard).
__device__ __forceinline__ void gbar(unsigned* cnt, unsigned& ep) {
  __syncthreads();
  __threadfence();
  ++ep;
  if (threadIdx.x == 0) {
    __hip_atomic_fetch_add(cnt, 1u, __ATOMIC_RELEASE, __HIP_MEMORY_SCOPE_AGENT);
    const unsigned tgt = ep * (unsigned)NWG_;
    while (__hip_atomic_load(cnt, __ATOMIC_ACQUIRE, __HIP_MEMORY_SCOPE_AGENT) < tgt)
      __builtin_amdgcn_s_sleep(2);
  }
  __syncthreads();
  __threadfence();
}

__global__ void k_init(float* h, float* c, unsigned* cnt) {
  const size_t i = (size_t)blockIdx.x * blockDim.x + threadIdx.x;
  const size_t n = (size_t)NB_ * B_ * H_;
  if (i < n) { h[i] = 0.0f; c[i] = 0.0f; }
  if (i == 0) *cnt = 0u;
}

// Convert the four weight tensors (each NB*4H*H = 2^24 fp32) into one
// contiguous bf16 region: [Wih | Whh | Wfc | Wproj].
__global__ void k_convert(const float* __restrict__ Wih, const float* __restrict__ Whh,
                          const float* __restrict__ Wfc, const float* __restrict__ Wpr,
                          u16* __restrict__ dst) {
  const float* srcs[4] = { Wih, Whh, Wfc, Wpr };
  const size_t gid = (size_t)blockIdx.x * blockDim.x + threadIdx.x;
  const size_t e   = gid * 4;                     // 4 elems/thread, same region
  const int    r   = (int)(e >> 24);
  const size_t w   = e & ((1u << 24) - 1u);
  const f32x4  f   = *(const f32x4*)(srcs[r] + w);
  u16x4 o; o.x = f2bf(f.x); o.y = f2bf(f.y); o.z = f2bf(f.z); o.w = f2bf(f.w);
  *(u16x4*)(dst + e) = o;
}

__global__ void __launch_bounds__(NTHR_, 1) k_lstm(Params P) {
  const int tid  = threadIdx.x;
  const int lane = tid & 31;
  const int gw   = blockIdx.x * (NTHR_ / 32) + (tid >> 5);   // 0..1023 waves
  const int b    = blockIdx.x;                               // batch row for elementwise phases
  __shared__ float r1[NTHR_], r2[NTHR_];
  unsigned ep = 0;
  const size_t BH = (size_t)B_ * H_;

  for (int t = 0; t < S_; ++t) {
    for (int k = 0; k < NB_; ++k) {
      // ------- Phase A: (gather | proj-partial combine) + LN1 + bf16 casts ---
      {
        float xv[2];
        const int bt = b * S_ + t;
        if (k == 0) {
#pragma unroll
          for (int i = 0; i < 2; ++i) {
            const int j = tid + NTHR_ * i;
            float v;
            if      (j <  64) v = P.stab[P.stage[bt] * 64  +  j];
            else if (j < 128) v = P.ctab[P.egoc[bt]  * 64  + (j - 64)];
            else if (j < 192) v = P.ctab[P.oppc[bt]  * 64  + (j - 128)];
            else if (j < 320) v = P.atab[P.egoa[bt]  * 128 + (j - 192)];
            else if (j < 448) v = P.atab[P.oppa[bt]  * 128 + (j - 320)];
            else              v = P.gst[(size_t)bt * 576 + (j - 448)];
            xv[i] = v;
            P.x[b * H_ + j] = v;
          }
        } else {
          // combine proj partials deterministically: x = y + p0 + p1 + p2 + p3
#pragma unroll
          for (int i = 0; i < 2; ++i) {
            const int j = tid + NTHR_ * i;
            const size_t o = (size_t)b * H_ + j;
            float v = P.x[o];
            v += P.part[o];
            v += P.part[BH + o];
            v += P.part[2 * BH + o];
            v += P.part[3 * BH + o];
            P.x[o] = v;
            xv[i] = v;
          }
        }
        float s = 0.0f, ss = 0.0f;
#pragma unroll
        for (int i = 0; i < 2; ++i) { s += xv[i]; ss += xv[i] * xv[i]; }
        r1[tid] = s; r2[tid] = ss; __syncthreads();
        for (int o = NTHR_ / 2; o > 0; o >>= 1) {
          if (tid < o) { r1[tid] += r1[tid + o]; r2[tid] += r2[tid + o]; }
          __syncthreads();
        }
        const float mean = r1[0] * (1.0f / H_);
        const float var  = r2[0] * (1.0f / H_) - mean * mean;
        const float rstd = rsqrtf(var + 1e-5f);
#pragma unroll
        for (int i = 0; i < 2; ++i) {
          const int j = tid + NTHR_ * i;
          const float xn = (xv[i] - mean) * rstd * P.ln1g[k * H_ + j] + P.ln1b[k * H_ + j];
          P.xnbf[b * H_ + j] = f2bf(xn);
          P.hbf [b * H_ + j] = f2bf(P.h[((size_t)k * B_ + b) * H_ + j]);
        }
      }
      gbar(P.cnt, ep);
      // ------- Phase B: gates = xn@Wih^T + h@Whh^T + b  (1024 tiles/1024 waves)
      {
        const u16* Wih = P.Wih + (size_t)k * H4_ * H_;
        const u16* Whh = P.Whh + (size_t)k * H4_ * H_;
        const int m0 = (gw & 3) * 16;
        const int n0 = (gw >> 2) * 16;
        v8f acc = {};
        gemm_k(acc, P.xnbf, H_, Wih, H_, m0, n0, 0, H_, lane);
        gemm_k(acc, P.hbf,  H_, Whh, H_, m0, n0, 0, H_, lane);
        const int n  = n0 + (lane & 15);
        const int mb = m0 + (lane >> 4) * 8;
        const float bias = P.bih[k * H4_ + n] + P.bhh[k * H4_ + n];
#pragma unroll
        for (int v = 0; v < 8; ++v)
          P.gates[(size_t)(mb + v) * H4_ + n] = acc[v] + bias;
      }
      gbar(P.cnt, ep);
      // ------- Phase C: LSTM cell + residual + LN2 ---------------------------
      {
        float yv[2];
#pragma unroll
        for (int i = 0; i < 2; ++i) {
          const int j = tid + NTHR_ * i;
          const float gi = P.gates[(size_t)b * H4_ +           j];
          const float gf = P.gates[(size_t)b * H4_ + H_      + j];
          const float gg = P.gates[(size_t)b * H4_ + 2 * H_  + j];
          const float go = P.gates[(size_t)b * H4_ + 3 * H_  + j];
          const size_t ho = ((size_t)k * B_ + b) * H_ + j;
          const float cn = sigf(gf) * P.c[ho] + sigf(gi) * tanhf(gg);
          const float hn = sigf(go) * tanhf(cn);
          P.c[ho] = cn; P.h[ho] = hn;
          const float y = P.x[b * H_ + j] + hn;
          P.x[b * H_ + j] = y;
          yv[i] = y;
        }
        float s = 0.0f, ss = 0.0f;
#pragma unroll
        for (int i = 0; i < 2; ++i) { s += yv[i]; ss += yv[i] * yv[i]; }
        r1[tid] = s; r2[tid] = ss; __syncthreads();
        for (int o = NTHR_ / 2; o > 0; o >>= 1) {
          if (tid < o) { r1[tid] += r1[tid + o]; r2[tid] += r2[tid + o]; }
          __syncthreads();
        }
        const float mean = r1[0] * (1.0f / H_);
        const float var  = r2[0] * (1.0f / H_) - mean * mean;
        const float rstd = rsqrtf(var + 1e-5f);
#pragma unroll
        for (int i = 0; i < 2; ++i) {
          const int j = tid + NTHR_ * i;
          const float yn = (yv[i] - mean) * rstd * P.ln2g[k * H_ + j] + P.ln2b[k * H_ + j];
          P.xnbf[b * H_ + j] = f2bf(yn);
        }
      }
      gbar(P.cnt, ep);
      // ------- Phase D: fc = gelu(yn@Wfc^T + bfc) (bf16) ---------------------
      {
        const u16* Wfc = P.Wfc + (size_t)k * H4_ * H_;
        const int m0 = (gw & 3) * 16;
        const int n0 = (gw >> 2) * 16;
        v8f acc = {};
        gemm_k(acc, P.xnbf, H_, Wfc, H_, m0, n0, 0, H_, lane);
        const int n  = n0 + (lane & 15);
        const int mb = m0 + (lane >> 4) * 8;
        const float bias = P.bfc[k * H4_ + n];
#pragma unroll
        for (int v = 0; v < 8; ++v)
          P.fcbf[(size_t)(mb + v) * H4_ + n] = f2bf(geluf(acc[v] + bias));
      }
      gbar(P.cnt, ep);
      // ------- Phase E: proj partials, 4-way K-split (1024 jobs/1024 waves) ---
      {
        const u16* Wp = P.Wpr + (size_t)k * H_ * H4_;
        const int tile = gw & 255;
        const int ks   = gw >> 8;                 // K slice 0..3
        const int m0 = (tile & 3) * 16;
        const int n0 = (tile >> 2) * 16;
        v8f acc = {};
        gemm_k(acc, P.fcbf, H4_, Wp, H4_, m0, n0, ks * 1024, 1024, lane);
        const int n  = n0 + (lane & 15);
        const int mb = m0 + (lane >> 4) * 8;
        const float bias = (ks == 0) ? P.bproj[k * H_ + n] : 0.0f;
        float* dst = P.part + (size_t)ks * BH;
#pragma unroll
        for (int v = 0; v < 8; ++v)
          dst[(size_t)(mb + v) * H_ + n] = acc[v] + bias;
      }
      gbar(P.cnt, ep);
    }
  }
  // Final output: last block-step's x = y + p0+p1+p2+p3, [B,1,H] fp32.
  {
    const int idx = blockIdx.x * NTHR_ + tid;            // 0..32767
#pragma unroll
    for (int i = 0; i < 2; ++i) {
      const int j = idx + NWG_ * NTHR_ * i;              // 0..65535
      float v = P.x[j];
      v += P.part[j];
      v += P.part[BH + j];
      v += P.part[2 * BH + j];
      v += P.part[3 * BH + j];
      P.out[j] = v;
    }
  }
}

extern "C" void kernel_launch(void* const* d_in, const int* in_sizes, int n_in,
                              void* d_out, int out_size, void* d_ws, size_t ws_size,
                              hipStream_t stream) {
  (void)in_sizes; (void)n_in; (void)out_size; (void)ws_size;
  const int*   stage = (const int*)  d_in[0];
  const int*   egoc  = (const int*)  d_in[1];
  const int*   oppc  = (const int*)  d_in[2];
  const int*   egoa  = (const int*)  d_in[3];
  const int*   oppa  = (const int*)  d_in[4];
  const float* gst   = (const float*)d_in[5];
  const float* stab  = (const float*)d_in[6];
  const float* ctab  = (const float*)d_in[7];
  const float* atab  = (const float*)d_in[8];
  const float* ln1g  = (const float*)d_in[9];
  const float* ln1b  = (const float*)d_in[10];
  const float* Wihf  = (const float*)d_in[11];
  const float* Whhf  = (const float*)d_in[12];
  const float* bih   = (const float*)d_in[13];
  const float* bhh   = (const float*)d_in[14];
  const float* ln2g  = (const float*)d_in[15];
  const float* ln2b  = (const float*)d_in[16];
  const float* Wfcf  = (const float*)d_in[17];
  const float* bfc   = (const float*)d_in[18];
  const float* Wprf  = (const float*)d_in[19];
  const float* bproj = (const float*)d_in[20];

  // Workspace layout (256B aligned). bf16 weights: 4 * 2^24 * 2B = 134.2 MB
  // (L2-resident on 192MB L2); activations/state/partials ~5.2 MB.
  char* ws = (char*)d_ws;
  size_t off = 0;
  auto take = [&](size_t bytes) -> char* {
    char* p = ws + off;
    off = (off + bytes + 255) & ~(size_t)255;
    return p;
  };
  const size_t WELEM = (size_t)NB_ * H4_ * H_;          // 2^24 per tensor
  u16*      wbf   = (u16*)     take(4 * WELEM * sizeof(u16));
  float*    x     = (float*)   take((size_t)B_ * H_ * 4);
  u16*      xnbf  = (u16*)     take((size_t)B_ * H_ * 2);
  u16*      hbf   = (u16*)     take((size_t)B_ * H_ * 2);
  float*    h     = (float*)   take((size_t)NB_ * B_ * H_ * 4);
  float*    c     = (float*)   take((size_t)NB_ * B_ * H_ * 4);
  float*    gates = (float*)   take((size_t)B_ * H4_ * 4);
  u16*      fcbf  = (u16*)     take((size_t)B_ * H4_ * 2);
  float*    part  = (float*)   take((size_t)4 * B_ * H_ * 4);
  unsigned* cnt   = (unsigned*)take(256);

  // 1) zero h/c state + barrier counter (deterministic per launch)
  k_init<<<(NB_ * B_ * H_ + 255) / 256, 256, 0, stream>>>(h, c, cnt);
  // 2) one-time fp32->bf16 weight conversion (4 * 2^24 elems, 4 per thread)
  k_convert<<<(unsigned)(4 * WELEM / (256 * 4)), 256, 0, stream>>>(Wihf, Whhf, Wfcf, Wprf, wbf);
  // 3) persistent fused LSTM-stack kernel
  Params P;
  P.stage = stage; P.egoc = egoc; P.oppc = oppc; P.egoa = egoa; P.oppa = oppa;
  P.gst = gst; P.stab = stab; P.ctab = ctab; P.atab = atab;
  P.ln1g = ln1g; P.ln1b = ln1b; P.bih = bih; P.bhh = bhh;
  P.ln2g = ln2g; P.ln2b = ln2b; P.bfc = bfc; P.bproj = bproj;
  P.Wih = wbf;             P.Whh = wbf + WELEM;
  P.Wfc = wbf + 2 * WELEM; P.Wpr = wbf + 3 * WELEM;
  P.x = x; P.h = h; P.c = c; P.gates = gates; P.part = part; P.out = (float*)d_out;
  P.xnbf = xnbf; P.hbf = hbf; P.fcbf = fcbf; P.cnt = cnt;
  k_lstm<<<NWG_, NTHR_, 0, stream>>>(P);
}
